// SurvPath_6640019439896
// MI455X (gfx1250) — matrix operand order repaired and checked by reading, not compiled, not run on hardware
//
#include <hip/hip_runtime.h>
#include <hip/hip_bf16.h>
#include <math.h>

typedef __attribute__((ext_vector_type(16))) __bf16 v16bf;
typedef __attribute__((ext_vector_type(8)))  float  v8f;

#define NROWS 50000
#define NT    3125          // 50000 / 16 row-tiles
#define NBLK  512           // persistent blocks for kernel B

// ---- workspace layout (float indices unless noted) ----
#define WS_QP   0           // 128 floats: q_p * scale
#define WS_VP   128         // 128 floats: v_p
#define WS_E0   256         // 1 float: exp(s0)
#define WS_PART 512         // NBLK * 132 floats of per-block partials
#define PART_STRIDE 132
#define OFF_PACKP  272384u  // byte offset: Wp packed bf16 (1024x256 -> 512KB)
#define OFF_PACKKV 796672u  // byte offset: Wkv packed bf16 (256x256 -> 128KB)

// ---- kernel B LDS layout (byte offsets) ----
#define R_A    0            // A tile bf16 16 x 1024, row stride 2064 B (pad)
#define A_STR  2064
#define R_E    33024        // embed f32 16 x 264 (stride 264 floats)
#define E_STR  264
#define R_XN   49920        // layernormed bf16 16 x 264 (stride 264 bf16 = 528 B)
#define XN_STR 528
#define R_RED  58368        // 16x16 f32
#define R_RED2 59392        // 16x16 f32
#define R_MU   60416
#define R_INV  60480
#define R_SME  60544        // exp(s) per row (16 f32)
#define R_QPS  60608        // q_p cached (128 f32)
#define R_ACC  61120        // block value accumulator (128 f32)
#define R_L    61632        // block sum-exp (1 f32)
#define SMEM_B 61696

__device__ __forceinline__ unsigned short f2bf(float x) {
    union { __bf16 h; unsigned short u; } c; c.h = (__bf16)x; return c.u;
}
__device__ __forceinline__ float elu1(float x)  { return x > 0.f ? x : (expf(x) - 1.f); }
__device__ __forceinline__ float gelu_exact(float x) {
    return 0.5f * x * (1.0f + erff(x * 0.70710678118654752f));
}

union FragBF { v16bf v; float4 q[2]; };
union Acc8   { v8f  v; float  f[8]; };

// -------------------- kernel P: pack Wp and Wkv into WMMA-B bf16 fragment order --------------------
// B-matrix (KxN) fragment order used here (contiguous k, same as A): lane(0..15)=col N, elems e=0..15 -> K=e;
// lane(16..31)=col N(lane-16), e -> K=16+e.  Packed addr: ((ntile*nKc + kc)*32 + lane)*16 + e.
__global__ void k_pack(const float* __restrict__ Wp, const float* __restrict__ Wqkv,
                       unsigned char* __restrict__ ws8) {
    const int NP1 = 1024 * 256;                 // Wp elements
    const int NP2 = 256 * 256;                  // Wkv elements
    int p = blockIdx.x * blockDim.x + threadIdx.x;
    if (p >= NP1 + NP2) return;
    bool m2 = (p >= NP1);
    int q = m2 ? (p - NP1) : p;
    int nKc = m2 ? 8 : 32;                      // K/32
    int e    = q & 15;
    int lane = (q >> 4) & 31;
    int rest = q >> 9;
    int kc    = rest % nKc;
    int ntile = rest / nKc;
    int n = ntile * 16 + (lane & 15);
    int klocal = (lane & 16) ? (16 + e) : e;
    int k = kc * 32 + klocal;
    float src = m2 ? Wqkv[k * 384 + 128 + n]    // columns 128..383 = K|V
                   : Wp[k * 256 + n];
    unsigned short* dst = (unsigned short*)(ws8 + (m2 ? OFF_PACKKV : OFF_PACKP));
    dst[q] = f2bf(src);
}

// -------------------- kernel A: omic MLP + token-0 qkv --------------------
__global__ void k_omic(const float* __restrict__ x_omic,
                       const float* __restrict__ W1, const float* __restrict__ b1,
                       const float* __restrict__ W2, const float* __restrict__ b2,
                       const float* __restrict__ ln1g, const float* __restrict__ ln1b,
                       const float* __restrict__ Wqkv, float* __restrict__ ws) {
    __shared__ float xs[1600], h1[256], h2[256], xn[256], qkv[384], st[4];
    int t = threadIdx.x;
    for (int i = t; i < 1577; i += 256) xs[i] = x_omic[i];
    __syncthreads();
    { float s = b1[t]; for (int i = 0; i < 1577; ++i) s += xs[i] * W1[i * 256 + t]; h1[t] = elu1(s); }
    __syncthreads();
    { float s = b2[t]; for (int i = 0; i < 256; ++i) s += h1[i] * W2[i * 256 + t]; h2[t] = elu1(s); }
    __syncthreads();
    if (t == 0) {
        float mu = 0.f, s2 = 0.f;
        for (int i = 0; i < 256; ++i) mu += h2[i];
        mu *= (1.f / 256.f);
        for (int i = 0; i < 256; ++i) { float d = h2[i] - mu; s2 += d * d; }
        st[0] = mu; st[1] = rsqrtf(s2 * (1.f / 256.f) + 1e-5f);
    }
    __syncthreads();
    xn[t] = (h2[t] - st[0]) * st[1] * ln1g[t] + ln1b[t];
    __syncthreads();
    for (int j = t; j < 384; j += 256) {
        float s = 0.f;
        for (int i = 0; i < 256; ++i) s += xn[i] * Wqkv[i * 384 + j];
        qkv[j] = s;
    }
    __syncthreads();
    const float scale = 0.0883883476483184f;   // 128^-0.5
    if (t < 128) { ws[WS_QP + t] = qkv[t] * scale; ws[WS_VP + t] = qkv[256 + t]; }
    __syncthreads();
    if (t == 0) {
        float s0 = 0.f;
        for (int n = 0; n < 128; ++n) s0 += (qkv[n] * scale) * qkv[128 + n];
        ws[WS_E0] = expf(s0);
    }
}

// -------------------- kernel B: fused streaming embed->LN->kv->attention accumulate --------------------
__global__ void __launch_bounds__(256) k_stream(const float* __restrict__ wsi,
                                                const float* __restrict__ bp,
                                                const float* __restrict__ ln1g,
                                                const float* __restrict__ ln1b,
                                                float* __restrict__ ws) {
    __shared__ __align__(16) unsigned char smem[SMEM_B];
    unsigned char* ws8 = (unsigned char*)ws;
    const unsigned char* packP  = ws8 + OFF_PACKP;
    const unsigned char* packKV = ws8 + OFF_PACKKV;

    const int t    = threadIdx.x;
    const int lane = t & 31;
    const int wid  = t >> 5;              // 8 waves
    float* accB = (float*)(smem + R_ACC);
    float* lB   = (float*)(smem + R_L);
    float* qps  = (float*)(smem + R_QPS);

    if (t < 128) { accB[t] = 0.f; qps[t] = ws[WS_QP + t]; }
    if (t == 128) lB[0] = 0.f;

    for (int tile = blockIdx.x; tile < NT; tile += NBLK) {
        __syncthreads();                               // protect R_A (KV) reuse across iterations
        // ---- stage A tile: 16 x 1024 fp32 -> bf16 LDS ----
        const float* wsrc = wsi + (size_t)tile * 16384;
        #pragma unroll
        for (int i = 0; i < 16; ++i) {
            int idx4 = t + i * 256;                    // 0..4095 float4s
            float4 w = ((const float4*)wsrc)[idx4];
            int f = idx4 << 2; int row = f >> 10; int col = f & 1023;
            uint2 pk;
            pk.x = (unsigned)f2bf(w.x) | ((unsigned)f2bf(w.y) << 16);
            pk.y = (unsigned)f2bf(w.z) | ((unsigned)f2bf(w.w) << 16);
            *(uint2*)(smem + R_A + row * A_STR + col * 2) = pk;
        }
        if (tile + NBLK < NT)                          // prefetch next tile (global_prefetch_b8)
            __builtin_prefetch((const char*)(wsi + (size_t)(tile + NBLK) * 16384) + t * 256, 0, 1);
        __syncthreads();

        // ---- GEMM1: E(16x256) = A(16x1024) x Wp ----
        Acc8 a0, a1;
        #pragma unroll
        for (int e = 0; e < 8; ++e) { a0.f[e] = 0.f; a1.f[e] = 0.f; }
        const int n0 = wid * 2, n1 = n0 + 1;
        const unsigned char* arow = smem + R_A + (lane & 15) * A_STR;
        for (int kc = 0; kc < 32; ++kc) {
            int kb = kc * 32 + ((lane & 16) ? 16 : 0); // A layout: {0..15} / {16..31}
            FragBF fa;
            fa.q[0] = *(const float4*)(arow + kb * 2);
            fa.q[1] = *(const float4*)(arow + (kb + 8) * 2);
            FragBF fb0, fb1;
            const float4* p0 = (const float4*)(packP + ((unsigned)((n0 * 32 + kc) * 32 + lane) << 5));
            const float4* p1 = (const float4*)(packP + ((unsigned)((n1 * 32 + kc) * 32 + lane) << 5));
            fb0.q[0] = p0[0]; fb0.q[1] = p0[1];
            fb1.q[0] = p1[0]; fb1.q[1] = p1[1];
            a0.v = __builtin_amdgcn_wmma_f32_16x16x32_bf16(false, fa.v, false, fb0.v, (short)0, a0.v, false, false);
            a1.v = __builtin_amdgcn_wmma_f32_16x16x32_bf16(false, fa.v, false, fb1.v, (short)0, a1.v, false, false);
        }
        // ---- write E + bias (C layout: VGPR e -> row (lane<16?e:8+e), col = lane%16 of ntile) ----
        float* Ef = (float*)(smem + R_E);
        {
            int rb = (lane & 16) ? 8 : 0;
            int c0 = n0 * 16 + (lane & 15), c1 = n1 * 16 + (lane & 15);
            float bb0 = bp[c0], bb1 = bp[c1];
            #pragma unroll
            for (int e = 0; e < 8; ++e) {
                Ef[(rb + e) * E_STR + c0] = a0.f[e] + bb0;
                Ef[(rb + e) * E_STR + c1] = a1.f[e] + bb1;
            }
        }
        __syncthreads();

        // ---- layernorm rows of E -> XN (bf16) ----
        float* red  = (float*)(smem + R_RED);
        float* red2 = (float*)(smem + R_RED2);
        {
            int row = t & 15, seg = t >> 4;            // seg covers 16 cols
            float s = 0.f, s2 = 0.f;
            #pragma unroll
            for (int c = 0; c < 16; ++c) { float x = Ef[row * E_STR + seg * 16 + c]; s += x; s2 += x * x; }
            red[row * 16 + seg] = s; red2[row * 16 + seg] = s2;
        }
        __syncthreads();
        if (t < 16) {
            float s = 0.f, s2 = 0.f;
            for (int g = 0; g < 16; ++g) { s += red[t * 16 + g]; s2 += red2[t * 16 + g]; }
            float mu = s * (1.f / 256.f);
            float var = s2 * (1.f / 256.f) - mu * mu;
            ((float*)(smem + R_MU))[t]  = mu;
            ((float*)(smem + R_INV))[t] = rsqrtf(var + 1e-5f);
        }
        __syncthreads();
        {
            int row = t & 15, seg = t >> 4;
            float mu = ((float*)(smem + R_MU))[row], inv = ((float*)(smem + R_INV))[row];
            unsigned char* xnrow = smem + R_XN + row * XN_STR;
            #pragma unroll
            for (int cc = 0; cc < 8; ++cc) {
                int c = seg * 16 + 2 * cc;
                float v0 = (Ef[row * E_STR + c]     - mu) * inv * ln1g[c]     + ln1b[c];
                float v1 = (Ef[row * E_STR + c + 1] - mu) * inv * ln1g[c + 1] + ln1b[c + 1];
                *(unsigned*)(xnrow + c * 2) = (unsigned)f2bf(v0) | ((unsigned)f2bf(v1) << 16);
            }
        }
        __syncthreads();

        // ---- GEMM2: KV(16x256) = XN(16x256) x Wkv ----
        Acc8 k0, k1;
        #pragma unroll
        for (int e = 0; e < 8; ++e) { k0.f[e] = 0.f; k1.f[e] = 0.f; }
        const unsigned char* xrow = smem + R_XN + (lane & 15) * XN_STR;
        for (int kc = 0; kc < 8; ++kc) {
            int kb = kc * 32 + ((lane & 16) ? 16 : 0);
            FragBF fa;
            fa.q[0] = *(const float4*)(xrow + kb * 2);
            fa.q[1] = *(const float4*)(xrow + (kb + 8) * 2);
            FragBF fb0, fb1;
            const float4* p0 = (const float4*)(packKV + ((unsigned)((n0 * 8 + kc) * 32 + lane) << 5));
            const float4* p1 = (const float4*)(packKV + ((unsigned)((n1 * 8 + kc) * 32 + lane) << 5));
            fb0.q[0] = p0[0]; fb0.q[1] = p0[1];
            fb1.q[0] = p1[0]; fb1.q[1] = p1[1];
            k0.v = __builtin_amdgcn_wmma_f32_16x16x32_bf16(false, fa.v, false, fb0.v, (short)0, k0.v, false, false);
            k1.v = __builtin_amdgcn_wmma_f32_16x16x32_bf16(false, fa.v, false, fb1.v, (short)0, k1.v, false, false);
        }
        float* KVf = (float*)(smem + R_A);             // overlay A region, stride 264 floats
        {
            int rb = (lane & 16) ? 8 : 0;
            int c0 = n0 * 16 + (lane & 15), c1 = n1 * 16 + (lane & 15);
            #pragma unroll
            for (int e = 0; e < 8; ++e) {
                KVf[(rb + e) * E_STR + c0] = k0.f[e];
                KVf[(rb + e) * E_STR + c1] = k1.f[e];
            }
        }
        __syncthreads();

        // ---- attention: s_j = qp . k_j ; accumulate exp(s_j), exp(s_j)*v_j ----
        {
            int row = t & 15, seg = t >> 4;            // seg covers 8 of 128 k dims
            float s = 0.f;
            #pragma unroll
            for (int c = 0; c < 8; ++c) s += qps[seg * 8 + c] * KVf[row * E_STR + seg * 8 + c];
            red[row * 16 + seg] = s;
        }
        __syncthreads();
        if (t < 16) {
            float s = 0.f;
            for (int g = 0; g < 16; ++g) s += red[t * 16 + g];
            ((float*)(smem + R_SME))[t] = expf(s);
        }
        __syncthreads();
        if (t < 128) {
            float a = 0.f;
            #pragma unroll
            for (int r = 0; r < 16; ++r) a += ((float*)(smem + R_SME))[r] * KVf[r * E_STR + 128 + t];
            accB[t] += a;
        } else if (t == 128) {
            float l = 0.f;
            for (int r = 0; r < 16; ++r) l += ((float*)(smem + R_SME))[r];
            lB[0] += l;
        }
    }
    __syncthreads();
    float* part = ws + WS_PART + blockIdx.x * PART_STRIDE;
    if (t < 128) part[t] = accB[t];
    if (t == 128) part[128] = lB[0];
}

// -------------------- kernel C: reduce + FFN epilogue + head --------------------
__global__ void k_final(const float* __restrict__ ln2g, const float* __restrict__ ln2b,
                        const float* __restrict__ Wf1, const float* __restrict__ bf1,
                        const float* __restrict__ Wf2, const float* __restrict__ bf2,
                        const float* __restrict__ ln3g, const float* __restrict__ ln3b,
                        const float* __restrict__ Wl1, const float* __restrict__ bl1,
                        const float* __restrict__ Wl2, const float* __restrict__ bl2,
                        const float* __restrict__ ws, float* __restrict__ out) {
    __shared__ float rowv[128], sx[128], su[128], sy[128], emb[256], hid[64], st[4], lg[4];
    int t = threadIdx.x;                               // 128 threads
    const float* part = ws + WS_PART;
    float e0 = ws[WS_E0];
    if (t == 0) {
        float l = e0;
        for (int b = 0; b < NBLK; ++b) l += part[b * PART_STRIDE + 128];
        st[0] = l;
    }
    __syncthreads();
    float acc = e0 * ws[WS_VP + t];
    for (int b = 0; b < NBLK; ++b) acc += part[b * PART_STRIDE + t];
    float outP = acc / st[0];
    float rowH = ws[WS_VP + t];                        // all histology rows collapse to v_p

    for (int r = 0; r < 2; ++r) {
        rowv[t] = (r == 0) ? outP : rowH;
        __syncthreads();
        if (t == 0) {                                  // LN2 stats
            float mu = 0.f, s2 = 0.f;
            for (int i = 0; i < 128; ++i) mu += rowv[i];
            mu *= (1.f / 128.f);
            for (int i = 0; i < 128; ++i) { float d = rowv[i] - mu; s2 += d * d; }
            st[1] = mu; st[2] = rsqrtf(s2 * (1.f / 128.f) + 1e-5f);
        }
        __syncthreads();
        sx[t] = (rowv[t] - st[1]) * st[2] * ln2g[t] + ln2b[t];
        __syncthreads();
        { float s = bf1[t]; for (int i = 0; i < 128; ++i) s += sx[i] * Wf1[i * 128 + t]; su[t] = gelu_exact(s); }
        __syncthreads();
        { float s = bf2[t]; for (int i = 0; i < 128; ++i) s += su[i] * Wf2[i * 128 + t]; sy[t] = s; }
        __syncthreads();
        if (t == 0) {                                  // LN3 stats
            float mu = 0.f, s2 = 0.f;
            for (int i = 0; i < 128; ++i) mu += sy[i];
            mu *= (1.f / 128.f);
            for (int i = 0; i < 128; ++i) { float d = sy[i] - mu; s2 += d * d; }
            st[1] = mu; st[2] = rsqrtf(s2 * (1.f / 128.f) + 1e-5f);
        }
        __syncthreads();
        emb[r * 128 + t] = (sy[t] - st[1]) * st[2] * ln3g[t] + ln3b[t];
        __syncthreads();
    }
    if (t < 64) {
        float s = bl1[t];
        for (int i = 0; i < 256; ++i) s += emb[i] * Wl1[i * 64 + t];
        hid[t] = fmaxf(s, 0.f);
    }
    __syncthreads();
    if (t < 4) {
        float s = bl2[t];
        for (int i = 0; i < 64; ++i) s += hid[i] * Wl2[i * 4 + t];
        lg[t] = s; out[t] = s;
    }
    __syncthreads();
    if (t == 0) {
        float m = fmaxf(fmaxf(lg[0], lg[1]), fmaxf(lg[2], lg[3]));
        float d = 0.f; float e[4];
        for (int i = 0; i < 4; ++i) { e[i] = expf(lg[i] - m); d += e[i]; }
        for (int i = 0; i < 4; ++i) out[4 + i] = e[i] / d;
    }
}

extern "C" void kernel_launch(void* const* d_in, const int* in_sizes, int n_in,
                              void* d_out, int out_size, void* d_ws, size_t ws_size,
                              hipStream_t stream) {
    (void)in_sizes; (void)n_in; (void)out_size; (void)ws_size;
    const float* wsi    = (const float*)d_in[0];
    const float* x_omic = (const float*)d_in[1];
    const float* W1     = (const float*)d_in[2];
    const float* b1     = (const float*)d_in[3];
    const float* W2     = (const float*)d_in[4];
    const float* b2     = (const float*)d_in[5];
    const float* Wp     = (const float*)d_in[6];
    const float* bp     = (const float*)d_in[7];
    const float* ln1g   = (const float*)d_in[8];
    const float* ln1b   = (const float*)d_in[9];
    const float* Wqkv   = (const float*)d_in[10];
    const float* ln2g   = (const float*)d_in[11];
    const float* ln2b   = (const float*)d_in[12];
    const float* Wf1    = (const float*)d_in[13];
    const float* bf1    = (const float*)d_in[14];
    const float* Wf2    = (const float*)d_in[15];
    const float* bf2    = (const float*)d_in[16];
    const float* ln3g   = (const float*)d_in[17];
    const float* ln3b   = (const float*)d_in[18];
    const float* Wl1    = (const float*)d_in[19];
    const float* bl1    = (const float*)d_in[20];
    const float* Wl2    = (const float*)d_in[21];
    const float* bl2    = (const float*)d_in[22];
    float* ws  = (float*)d_ws;
    float* out = (float*)d_out;

    const int NPACK = 1024 * 256 + 256 * 256;
    k_pack<<<(NPACK + 255) / 256, 256, 0, stream>>>(Wp, Wqkv, (unsigned char*)d_ws);
    k_omic<<<1, 256, 0, stream>>>(x_omic, W1, b1, W2, b2, ln1g, ln1b, Wqkv, ws);
    k_stream<<<NBLK, 256, 0, stream>>>(wsi, bp, ln1g, ln1b, ws);
    k_final<<<1, 128, 0, stream>>>(ln2g, ln2b, Wf1, bf1, Wf2, bf2, ln3g, ln3b,
                                   Wl1, bl1, Wl2, bl2, ws, out);
}
